// NormalizationLayer_11467562680485
// MI455X (gfx1250) — matrix operand order repaired and checked by reading, not compiled, not run on hardware
//
#include <hip/hip_runtime.h>
#include <hip/hip_bf16.h>

typedef __attribute__((ext_vector_type(16))) _Float16 v16h;
typedef __attribute__((ext_vector_type(8)))  _Float16 v8h;
typedef __attribute__((ext_vector_type(8)))  float    v8f;
typedef __attribute__((ext_vector_type(4)))  float    v4f;

constexpr int Bdim = 8;
constexpr int Ndim = 2048;
constexpr int Fdim = 256;
constexpr int ROWS = Bdim * Ndim;   // 16384 total rows across batches

union Frag16 {
    v16h v;
    v8h  h2[2];
    _Float16 h[16];
};

#define WMMA_F16(A, B, C) __builtin_amdgcn_wmma_f32_16x16x32_f16( \
    false, (A), false, (B), (short)0, (C), false, false)

// ---------------------------------------------------------------------------
// Kernel 1: c1 = v @ W1^T, c2 = v @ W2^T  (fp32 inputs, f16 outputs, f32 acc)
// One 16x16 output tile per wave; K=256 -> 8 x v_wmma_f32_16x16x32_f16.
// ---------------------------------------------------------------------------
__global__ __launch_bounds__(256)
void proj_kernel(const float* __restrict__ v,
                 const float* __restrict__ W1,
                 const float* __restrict__ W2,
                 _Float16* __restrict__ c1,
                 _Float16* __restrict__ c2) {
    const int lane    = threadIdx.x & 31;
    const int wt      = blockIdx.x * 8 + (threadIdx.x >> 5);
    const int mat     = wt >> 14;           // 16384 tiles per output matrix
    const int rem     = wt & 16383;
    const int row0    = (rem >> 4) << 4;    // 1024 row-tiles
    const int o0      = (rem & 15) << 4;    // 16 col-tiles
    const int halfsel = lane >> 4;          // 0: lanes 0-15, 1: lanes 16-31
    const int lr      = lane & 15;

    const float* __restrict__ W   = mat ? W2 : W1;
    _Float16*    __restrict__ dst = mat ? c2 : c1;

    const float* __restrict__ ap = v + (size_t)(row0 + lr) * Fdim + halfsel * 16;
    const float* __restrict__ bp = W + (size_t)(o0  + lr) * Fdim + halfsel * 16;

    v8f acc = {};
    #pragma unroll
    for (int k0 = 0; k0 < Fdim; k0 += 32) {
        Frag16 a, b;
        {   // A fragment: rows of v, 16 contiguous K {kb..kb+15}
            const v4f f0 = *(const v4f*)(ap + k0);
            const v4f f1 = *(const v4f*)(ap + k0 + 4);
            const v4f f2 = *(const v4f*)(ap + k0 + 8);
            const v4f f3 = *(const v4f*)(ap + k0 + 12);
            #pragma unroll
            for (int i = 0; i < 4; ++i) {
                a.h[i]      = (_Float16)f0[i];
                a.h[4 + i]  = (_Float16)f1[i];
                a.h[8 + i]  = (_Float16)f2[i];
                a.h[12 + i] = (_Float16)f3[i];
            }
        }
        {   // B fragment: row o of W = column o of W^T, 16 contiguous K
            const v4f f0 = *(const v4f*)(bp + k0);
            const v4f f1 = *(const v4f*)(bp + k0 + 4);
            const v4f f2 = *(const v4f*)(bp + k0 + 8);
            const v4f f3 = *(const v4f*)(bp + k0 + 12);
            #pragma unroll
            for (int i = 0; i < 4; ++i) {
                b.h[i]      = (_Float16)f0[i];
                b.h[4 + i]  = (_Float16)f1[i];
                b.h[8 + i]  = (_Float16)f2[i];
                b.h[12 + i] = (_Float16)f3[i];
            }
        }
        acc = WMMA_F16(a.v, b.v, acc);
    }

    // D layout: VGPR r, lane L -> (M = r + 8*(L>=16), N = L%16)
    #pragma unroll
    for (int r = 0; r < 8; ++r) {
        const int n = row0 + r + halfsel * 8;
        dst[(size_t)n * Fdim + o0 + lr] = (_Float16)acc[r];
    }
}

// ---------------------------------------------------------------------------
// Kernel 2: per batch, s = c1 @ c2^T with 32(M)x64(N) register blocking per
// wave: 2 A-frags x 4 B-frags -> 8 WMMAs per k-step, 8 f32 accumulators.
// Cuts c1/c2 fragment traffic from 64 B/output to 24 B/output so the WMMA
// pipes aren't starved by L1/L2.  adj tile is prefetched (global_prefetch)
// before the GEMM so the HBM fetch overlaps compute.  Fused epilogue:
// c = 100*sigmoid(s)+1e-5; cc = 1/(2c^2); out = exp(-(adj^2*cc)).
// ---------------------------------------------------------------------------
__global__ __launch_bounds__(256)
void score_kernel(const _Float16* __restrict__ c1,
                  const _Float16* __restrict__ c2,
                  const float* __restrict__ adj,
                  float* __restrict__ out) {
    const int lane    = threadIdx.x & 31;
    const int wt      = blockIdx.x * 8 + (threadIdx.x >> 5);
    // per batch: 64 row-tiles (32 rows) x 32 col-tiles (64 cols) = 2048 tiles
    const int b       = wt >> 11;
    const int rem     = wt & 2047;
    const int n0      = (rem >> 5) << 5;    // 0..2016 step 32
    const int m0      = (rem & 31) << 6;    // 0..1984 step 64
    const int halfsel = lane >> 4;
    const int lr      = lane & 15;

    const _Float16* __restrict__ base1 = c1 + (size_t)b * Ndim * Fdim;
    const _Float16* __restrict__ base2 = c2 + (size_t)b * Ndim * Fdim;
    const float*    __restrict__ adjb  = adj + (size_t)b * Ndim * Ndim;
    float*          __restrict__ outb  = out + (size_t)b * Ndim * Ndim;

    // Prefetch the 32x64 fp32 adj tile (8 KB): lane L covers row n0+L,
    // two 128B lines per row.  Overlaps the HBM fetch with the GEMM below.
    {
        const float* p = adjb + (size_t)(n0 + lane) * Ndim + m0;
        __builtin_prefetch(p, 0, 0);
        __builtin_prefetch(p + 32, 0, 0);
    }

    const _Float16* __restrict__ ap0 =
        base1 + (size_t)(n0 + lr) * Fdim + halfsel * 16;
    const _Float16* __restrict__ ap1 = ap0 + 16 * Fdim;
    const _Float16* __restrict__ bp0 =
        base2 + (size_t)(m0 + lr) * Fdim + halfsel * 16;
    const _Float16* __restrict__ bp1 = bp0 + 16 * Fdim;
    const _Float16* __restrict__ bp2 = bp0 + 32 * Fdim;
    const _Float16* __restrict__ bp3 = bp0 + 48 * Fdim;

    v8f acc[2][4] = {};
    #pragma unroll
    for (int k0 = 0; k0 < Fdim; k0 += 32) {
        Frag16 a0, a1, b0, b1, b2, b3;
        a0.h2[0] = *(const v8h*)(ap0 + k0);
        a0.h2[1] = *(const v8h*)(ap0 + k0 + 8);
        a1.h2[0] = *(const v8h*)(ap1 + k0);
        a1.h2[1] = *(const v8h*)(ap1 + k0 + 8);
        b0.v = *(const v16h*)(bp0 + k0);
        b1.v = *(const v16h*)(bp1 + k0);
        b2.v = *(const v16h*)(bp2 + k0);
        b3.v = *(const v16h*)(bp3 + k0);

        acc[0][0] = WMMA_F16(a0.v, b0.v, acc[0][0]);
        acc[0][1] = WMMA_F16(a0.v, b1.v, acc[0][1]);
        acc[0][2] = WMMA_F16(a0.v, b2.v, acc[0][2]);
        acc[0][3] = WMMA_F16(a0.v, b3.v, acc[0][3]);
        acc[1][0] = WMMA_F16(a1.v, b0.v, acc[1][0]);
        acc[1][1] = WMMA_F16(a1.v, b1.v, acc[1][1]);
        acc[1][2] = WMMA_F16(a1.v, b2.v, acc[1][2]);
        acc[1][3] = WMMA_F16(a1.v, b3.v, acc[1][3]);
    }

    // Fused epilogue over the 2x4 grid of 16x16 tiles.
    #pragma unroll
    for (int ti = 0; ti < 2; ++ti) {
        #pragma unroll
        for (int tj = 0; tj < 4; ++tj) {
            #pragma unroll
            for (int r = 0; r < 8; ++r) {
                const int n = n0 + ti * 16 + r + halfsel * 8;
                const int m = m0 + tj * 16 + lr;
                const size_t idx = (size_t)n * Ndim + m;
                const float s  = acc[ti][tj][r];
                const float c  = 100.0f * __frcp_rn(1.0f + __expf(-s)) + 1e-05f;
                const float cc = __frcp_rn(2.0f * c * c);
                const float a2 = adjb[idx];
                outb[idx] = __expf(-(a2 * a2 * cc));
            }
        }
    }
}

extern "C" void kernel_launch(void* const* d_in, const int* in_sizes, int n_in,
                              void* d_out, int out_size, void* d_ws, size_t ws_size,
                              hipStream_t stream) {
    const float* v   = (const float*)d_in[0];   // [8,2048,256]
    const float* adj = (const float*)d_in[1];   // [8,2048,2048]
    const float* W1  = (const float*)d_in[2];   // [256,256]
    const float* W2  = (const float*)d_in[3];   // [256,256]

    float* out = (float*)d_out;                 // [v | norm_adj] concatenated

    // Workspace: c1, c2 as f16 [8,2048,256] (8 MB each; L2-resident)
    _Float16* c1 = (_Float16*)d_ws;
    _Float16* c2 = c1 + (size_t)ROWS * Fdim;

    // Output 0: passthrough copy of v (16.8 MB, capture-safe async d2d)
    hipMemcpyAsync(out, v, (size_t)ROWS * Fdim * sizeof(float),
                   hipMemcpyDeviceToDevice, stream);

    // Stage 1: projections. 32768 tiles / 8 waves per block.
    proj_kernel<<<4096, 256, 0, stream>>>(v, W1, W2, c1, c2);

    // Stage 2: Gram matrix + fused normalization.
    // 16384 wave-tiles (32x64 each) / 8 waves per block.
    score_kernel<<<2048, 256, 0, stream>>>(c1, c2, adj,
                                           out + (size_t)ROWS * Fdim);
}
